// Qwen3VL2BMultimodalAttention_87273735455086
// MI455X (gfx1250) — matrix operand order repaired and checked by reading, not compiled, not run on hardware
//
#include <hip/hip_runtime.h>
#include <hip/hip_bf16.h>

// ---------------------------------------------------------------------------
// Types for CDNA5 WMMA (wave32, 16x16x32 bf16 -> f32) and TDM descriptors
// ---------------------------------------------------------------------------
typedef __attribute__((ext_vector_type(16))) __bf16        v16bf;
typedef __attribute__((ext_vector_type(8)))  float         v8f;
typedef __attribute__((ext_vector_type(4)))  unsigned int  v4u;
typedef __attribute__((ext_vector_type(8)))  int           v8i;
typedef __attribute__((ext_vector_type(4)))  int           v4i;

union FragAB { uint4 u[2]; v16bf v; };   // 32 bytes: one WMMA A or B operand

#define SEQ    2048
#define HID    2048
#define NH     16
#define NKV    8
#define HD     128
#define KVW    (NKV * HD)   // 1024

// ---------------------------------------------------------------------------
// Tensor Data Mover: 2-D tile load (bf16) from global -> LDS.
// Descriptor per CDNA5 ISA ch.8 (group0: count/lds/global/type, group1:
// data_size + tensor dims + tile dims + dim0 stride; groups 2/3 zero = 2D).
// LDS result is tile row-major contiguous.
// ---------------------------------------------------------------------------
__device__ inline void tdm_load_2d_bf16(const void* gptr, unsigned lds_addr,
                                        unsigned tile_d0, unsigned tile_d1,
                                        unsigned tensor_d0, unsigned tensor_d1,
                                        unsigned d0_stride /* elements */) {
    unsigned long long ga = (unsigned long long)gptr;
    v4u g0 = { 1u,                                   // count=1, user mode
               lds_addr,                             // LDS byte address
               (unsigned)ga,                         // global_addr[31:0]
               ((unsigned)(ga >> 32) & 0x01FFFFFFu)  // global_addr[56:32]
                   | (2u << 30) };                   // type=2 (image)
    v8i g1;
    g1[0] = (int)(1u << 16);                                  // data_size=1 (2B)
    g1[1] = (int)((tensor_d0 & 0xFFFFu) << 16);               // tensor_dim0[15:0]
    g1[2] = (int)((tensor_d0 >> 16) | ((tensor_d1 & 0xFFFFu) << 16));
    g1[3] = (int)((tensor_d1 >> 16) | (tile_d0 << 16));       // tile_dim0
    g1[4] = (int)(tile_d1);                                   // tile_dim1, tile_dim2=0
    g1[5] = (int)d0_stride;                                   // dim0 stride low32
    g1[6] = 0;                                                // stride hi, dim1 stride
    g1[7] = 0;
    v4i g2 = {0, 0, 0, 0};
    v4i g3 = {0, 0, 0, 0};
#if defined(__clang_major__) && (__clang_major__ >= 23)
    v8i g4 = {0, 0, 0, 0, 0, 0, 0, 0};
    __builtin_amdgcn_tensor_load_to_lds(g0, g1, g2, g3, g4, 0);
#else
    __builtin_amdgcn_tensor_load_to_lds(g0, g1, g2, g3, 0);
#endif
}

// ---------------------------------------------------------------------------
// fp32 -> bf16 elementwise conversion
// ---------------------------------------------------------------------------
__global__ void cvt_f32_bf16(const float* __restrict__ in, __bf16* __restrict__ out, int n) {
    int i = blockIdx.x * blockDim.x + threadIdx.x;
    int stride = gridDim.x * blockDim.x;
    for (; i < n; i += stride) out[i] = (__bf16)in[i];
}

// ---------------------------------------------------------------------------
// C[M,N] = A[M,K] x B[N,K]^T  (both bf16 row-major, K contiguous; f32 accum)
// Workgroup tile 128x128, BK=32, TDM-staged double-buffered LDS tiles.
// 8 waves in 2x4 grid, each wave 64x32 (4x2 WMMA 16x16 accumulators).
// ---------------------------------------------------------------------------
template<bool OUT_BF16>
__global__ __launch_bounds__(256)
void gemm_bf16(const __bf16* __restrict__ A, const __bf16* __restrict__ B,
               float* __restrict__ Cf, __bf16* __restrict__ Cb,
               int M, int N, int K) {
    __shared__ __bf16 As[2][128 * 32];
    __shared__ __bf16 Bs[2][128 * 32];

    const int tid  = threadIdx.x;
    const int lane = tid & 31;
    const int wave = tid >> 5;
    const int wm   = wave >> 2;      // 0..1
    const int wn   = wave & 3;       // 0..3
    const int hh   = lane >> 4;      // half-wave
    const int ll   = lane & 15;
    const int bm   = blockIdx.y * 128;
    const int bn   = blockIdx.x * 128;

    v8f acc[4][2];
#pragma unroll
    for (int i = 0; i < 4; ++i)
#pragma unroll
        for (int j = 0; j < 2; ++j)
#pragma unroll
            for (int e = 0; e < 8; ++e) acc[i][j][e] = 0.f;

    const int kTiles = K >> 5;

    // prologue: TDM prefetch of tile 0 (wave 0 issues, TENSORcnt-tracked)
    if (wave == 0) {
        tdm_load_2d_bf16(A + (size_t)bm * K, (unsigned)(unsigned long long)&As[0][0],
                         32, 128, (unsigned)K, (unsigned)M, (unsigned)K);
        tdm_load_2d_bf16(B + (size_t)bn * K, (unsigned)(unsigned long long)&Bs[0][0],
                         32, 128, (unsigned)K, (unsigned)N, (unsigned)K);
    }

    for (int kt = 0; kt < kTiles; ++kt) {
        const int cur = kt & 1;
        if (wave == 0) {
            if (kt + 1 < kTiles) {
                const int k1 = (kt + 1) << 5;
                tdm_load_2d_bf16(A + (size_t)bm * K + k1,
                                 (unsigned)(unsigned long long)&As[cur ^ 1][0],
                                 32, 128, (unsigned)K, (unsigned)M, (unsigned)K);
                tdm_load_2d_bf16(B + (size_t)bn * K + k1,
                                 (unsigned)(unsigned long long)&Bs[cur ^ 1][0],
                                 32, 128, (unsigned)K, (unsigned)N, (unsigned)K);
                __builtin_amdgcn_s_wait_tensorcnt((short)2);   // current tile done
            } else {
                __builtin_amdgcn_s_wait_tensorcnt((short)0);
            }
        }
        __syncthreads();   // tile `cur` visible to all waves

        // A fragments: lane half 0 -> K 0-7 & 16-23, half 1 -> K 8-15 & 24-31
        FragAB afr[4], bfr[2];
#pragma unroll
        for (int i = 0; i < 4; ++i) {
            int r  = wm * 64 + i * 16 + ll;
            int c0 = hh * 8;
            afr[i].u[0] = *(const uint4*)&As[cur][r * 32 + c0];
            afr[i].u[1] = *(const uint4*)&As[cur][r * 32 + c0 + 16];
        }
        // B fragments: lane half 0 -> K 0-7 & 16-23, half 1 -> K 8-15 & 24-31 of column n (= row of W)
#pragma unroll
        for (int j = 0; j < 2; ++j) {
            int n  = wn * 32 + j * 16 + ll;
            int c0 = hh * 8;
            bfr[j].u[0] = *(const uint4*)&Bs[cur][n * 32 + c0];
            bfr[j].u[1] = *(const uint4*)&Bs[cur][n * 32 + c0 + 16];
        }
#pragma unroll
        for (int i = 0; i < 4; ++i)
#pragma unroll
            for (int j = 0; j < 2; ++j)
                acc[i][j] = __builtin_amdgcn_wmma_f32_16x16x32_bf16(
                    false, afr[i].v, false, bfr[j].v, (short)0, acc[i][j], false, false);
        __syncthreads();   // all waves done with tile `cur` before TDM overwrites
    }

    // store: C tile layout VGPR v -> row v (lanes 0-15) / v+8 (lanes 16-31)
#pragma unroll
    for (int i = 0; i < 4; ++i)
#pragma unroll
        for (int j = 0; j < 2; ++j) {
            int n = bn + wn * 32 + j * 16 + ll;
#pragma unroll
            for (int v = 0; v < 8; ++v) {
                int m = bm + wm * 64 + i * 16 + v + hh * 8;
                float val = acc[i][j][v];
                if (OUT_BF16) Cb[(size_t)m * N + n] = (__bf16)val;
                else          Cf[(size_t)m * N + n] = val;
            }
        }
}

// ---------------------------------------------------------------------------
// Flash attention: grid (head, qblock). 8 waves, each owns 16 query rows.
// K block (128x128 bf16) staged via TDM; after S, same buffer reused for V^T.
// Online softmax in f32; P re-materialized via per-wave LDS for the PV WMMA.
// ---------------------------------------------------------------------------
__global__ __launch_bounds__(256)
void flash_attn(const __bf16* __restrict__ Qg, const __bf16* __restrict__ Kg,
                const __bf16* __restrict__ Vg, __bf16* __restrict__ Og) {
    __shared__ __bf16 Ks[128 * 128];       // [key][d] for K, then [d][key] for V^T
    __shared__ __bf16 Pw[8][16 * 128];     // per-wave P tile (bf16)

    const int h    = blockIdx.x;           // query head
    const int qb   = blockIdx.y;           // 128-row query block
    const int kvh  = h >> 1;               // GQA: repeat_interleave -> h / 2
    const int tid  = threadIdx.x;
    const int lane = tid & 31;
    const int wave = tid >> 5;
    const int hh   = lane >> 4;
    const int ll   = lane & 15;
    const int qrow = qb * 128 + wave * 16 + ll;

    const float c = 0.08838834764831845f * 1.4426950408889634f;  // scale * log2(e)

    // Q fragments for this wave's 16 rows: 4 frags along d=128 (K dim)
    FragAB qf[4];
#pragma unroll
    for (int df = 0; df < 4; ++df) {
        int c0 = df * 32 + hh * 8;
        qf[df].u[0] = *(const uint4*)&Qg[(size_t)qrow * HID + h * HD + c0];
        qf[df].u[1] = *(const uint4*)&Qg[(size_t)qrow * HID + h * HD + c0 + 16];
    }

    v8f   o[8];
    float mrow[8], lrow[8];
#pragma unroll
    for (int t = 0; t < 8; ++t)
#pragma unroll
        for (int e = 0; e < 8; ++e) o[t][e] = 0.f;
#pragma unroll
    for (int v = 0; v < 8; ++v) { mrow[v] = -1e30f; lrow[v] = 0.f; }

    for (int kb = 0; kb < 16; ++kb) {
        // stage K block [key][d] via Tensor Data Mover
        if (wave == 0) {
            tdm_load_2d_bf16(Kg + (size_t)(kb * 128) * KVW + kvh * HD,
                             (unsigned)(unsigned long long)&Ks[0],
                             128, 128, (unsigned)KVW, (unsigned)SEQ, (unsigned)KVW);
            __builtin_amdgcn_s_wait_tensorcnt((short)0);
        }
        __syncthreads();

        // S = Q x K^T : 8 tiles over 128 keys, 4 K-steps each
        v8f s[8];
#pragma unroll
        for (int t = 0; t < 8; ++t) {
            v8f z;
#pragma unroll
            for (int e = 0; e < 8; ++e) z[e] = 0.f;
#pragma unroll
            for (int df = 0; df < 4; ++df) {
                FragAB bf_;
                int key = t * 16 + ll;
                int d0  = df * 32 + hh * 8;
                bf_.u[0] = *(const uint4*)&Ks[key * 128 + d0];
                bf_.u[1] = *(const uint4*)&Ks[key * 128 + d0 + 16];
                z = __builtin_amdgcn_wmma_f32_16x16x32_bf16(
                    false, qf[df].v, false, bf_.v, (short)0, z, false, false);
            }
            s[t] = z;
        }

        // row max (VGPR v <-> row v + hh*8; reduce within half-wave)
        float mloc[8];
#pragma unroll
        for (int v = 0; v < 8; ++v) {
            float m = s[0][v];
#pragma unroll
            for (int t = 1; t < 8; ++t) m = fmaxf(m, s[t][v]);
#pragma unroll
            for (int msk = 1; msk < 16; msk <<= 1)
                m = fmaxf(m, __shfl_xor(m, msk, 32));
            mloc[v] = m;
        }

        float rs[8];
#pragma unroll
        for (int v = 0; v < 8; ++v) {
            float mnew = fmaxf(mrow[v], mloc[v]);
            rs[v]   = exp2f((mrow[v] - mnew) * c);
            mrow[v] = mnew;
        }
#pragma unroll
        for (int t = 0; t < 8; ++t)
#pragma unroll
            for (int v = 0; v < 8; ++v) o[t][v] *= rs[v];

        // P = exp2((s - m)*c); write bf16 to per-wave LDS; accumulate row sums
        float lad[8];
#pragma unroll
        for (int v = 0; v < 8; ++v) lad[v] = 0.f;
#pragma unroll
        for (int t = 0; t < 8; ++t)
#pragma unroll
            for (int v = 0; v < 8; ++v) {
                float p = exp2f((s[t][v] - mrow[v]) * c);
                lad[v] += p;
                Pw[wave][(v + hh * 8) * 128 + t * 16 + ll] = (__bf16)p;
            }
#pragma unroll
        for (int v = 0; v < 8; ++v) {
            float la = lad[v];
#pragma unroll
            for (int msk = 1; msk < 16; msk <<= 1) la += __shfl_xor(la, msk, 32);
            lrow[v] = lrow[v] * rs[v] + la;
        }
        __syncthreads();                 // all waves done reading K data

        // stage V transposed: Ks[d][key] so B-fragment columns are contiguous
#pragma unroll
        for (int cidx = tid; cidx < 2048; cidx += 256) {
            int key = cidx >> 4;
            int d0  = (cidx & 15) << 3;
            uint4 raw = *(const uint4*)&Vg[(size_t)(kb * 128 + key) * KVW + kvh * HD + d0];
            const __bf16* pv = (const __bf16*)&raw;
#pragma unroll
            for (int e = 0; e < 8; ++e) Ks[(d0 + e) * 128 + key] = pv[e];
        }
        __syncthreads();

        // O += P x V : A frags from Pw, B frags from V^T
#pragma unroll
        for (int kf = 0; kf < 4; ++kf) {
            FragAB pf;
            int c0 = kf * 32 + hh * 8;
            pf.u[0] = *(const uint4*)&Pw[wave][ll * 128 + c0];
            pf.u[1] = *(const uint4*)&Pw[wave][ll * 128 + c0 + 16];
#pragma unroll
            for (int t = 0; t < 8; ++t) {
                FragAB vf;
                int d  = t * 16 + ll;
                int k0 = kf * 32 + hh * 8;
                vf.u[0] = *(const uint4*)&Ks[d * 128 + k0];
                vf.u[1] = *(const uint4*)&Ks[d * 128 + k0 + 16];
                o[t] = __builtin_amdgcn_wmma_f32_16x16x32_bf16(
                    false, pf.v, false, vf.v, (short)0, o[t], false, false);
            }
        }
        __syncthreads();                 // before next K block overwrites Ks
    }

    // epilogue: normalize and store bf16 into attention-output activation
#pragma unroll
    for (int t = 0; t < 8; ++t)
#pragma unroll
        for (int v = 0; v < 8; ++v) {
            int m = qb * 128 + wave * 16 + v + hh * 8;
            int d = t * 16 + ll;
            Og[(size_t)m * HID + h * HD + d] = (__bf16)(o[t][v] / lrow[v]);
        }
}

// ---------------------------------------------------------------------------
// Launch
// ---------------------------------------------------------------------------
extern "C" void kernel_launch(void* const* d_in, const int* in_sizes, int n_in,
                              void* d_out, int out_size, void* d_ws, size_t ws_size,
                              hipStream_t stream) {
    const float* hidden = (const float*)d_in[0];
    const float* wq     = (const float*)d_in[1];
    const float* wk     = (const float*)d_in[2];
    const float* wv     = (const float*)d_in[3];
    const float* wo     = (const float*)d_in[4];
    float* out = (float*)d_out;

    char* ws = (char*)d_ws;
    size_t off = 0;
    __bf16* hid_b = (__bf16*)(ws + off); off += (size_t)SEQ * HID * 2;
    __bf16* wq_b  = (__bf16*)(ws + off); off += (size_t)HID * HID * 2;
    __bf16* wk_b  = (__bf16*)(ws + off); off += (size_t)KVW * HID * 2;
    __bf16* wv_b  = (__bf16*)(ws + off); off += (size_t)KVW * HID * 2;
    __bf16* wo_b  = (__bf16*)(ws + off); off += (size_t)HID * HID * 2;
    __bf16* q_b   = (__bf16*)(ws + off); off += (size_t)SEQ * HID * 2;
    __bf16* k_b   = (__bf16*)(ws + off); off += (size_t)SEQ * KVW * 2;
    __bf16* v_b   = (__bf16*)(ws + off); off += (size_t)SEQ * KVW * 2;
    __bf16* ao_b  = (__bf16*)(ws + off); off += (size_t)SEQ * HID * 2;

    const int nHid = SEQ * HID, nSq = HID * HID, nKv = KVW * HID;
    cvt_f32_bf16<<<2048, 256, 0, stream>>>(hidden, hid_b, nHid);
    cvt_f32_bf16<<<2048, 256, 0, stream>>>(wq, wq_b, nSq);
    cvt_f32_bf16<<<1024, 256, 0, stream>>>(wk, wk_b, nKv);
    cvt_f32_bf16<<<1024, 256, 0, stream>>>(wv, wv_b, nKv);
    cvt_f32_bf16<<<2048, 256, 0, stream>>>(wo, wo_b, nSq);

    // projections: x @ W^T
    gemm_bf16<true><<<dim3(HID / 128, SEQ / 128), 256, 0, stream>>>(
        hid_b, wq_b, nullptr, q_b, SEQ, HID, HID);
    gemm_bf16<true><<<dim3(KVW / 128, SEQ / 128), 256, 0, stream>>>(
        hid_b, wk_b, nullptr, k_b, SEQ, KVW, HID);
    gemm_bf16<true><<<dim3(KVW / 128, SEQ / 128), 256, 0, stream>>>(
        hid_b, wv_b, nullptr, v_b, SEQ, KVW, HID);

    // fused softmax(QK^T * scale) V, GQA 2:1
    flash_attn<<<dim3(NH, SEQ / 128), 256, 0, stream>>>(q_b, k_b, v_b, ao_b);

    // output projection, f32 result
    gemm_bf16<false><<<dim3(HID / 128, SEQ / 128), 256, 0, stream>>>(
        ao_b, wo_b, out, nullptr, SEQ, HID, HID);
}
